// Caspar_49512382988780
// MI455X (gfx1250) — matrix operand order repaired and checked by reading, not compiled, not run on hardware
//
#include <hip/hip_runtime.h>
#include <math.h>

#define SEQ 384
#define DIN 768
#define HD  1536
#define FF  768
#define NA  128

typedef __attribute__((ext_vector_type(16))) __bf16 v16bf;
typedef __attribute__((ext_vector_type(8)))  float  v8f;

static __device__ __forceinline__ __bf16 us2bf(unsigned short u) {
    union { unsigned short u; __bf16 h; } x; x.u = u; return x.h;
}
static __device__ __forceinline__ unsigned short f2bf_rne(float f) {
    union { float f; unsigned int u; } x; x.f = f;
    unsigned int r = ((x.u >> 16) & 1u) + 0x7FFFu;
    return (unsigned short)((x.u + r) >> 16);
}

// ---------------------------------------------------------------------------
// fp32 -> bf16 (round-nearest-even), straight copy
// ---------------------------------------------------------------------------
__global__ void Caspar_cvt_bf16(const float* __restrict__ in,
                                unsigned short* __restrict__ out, int n) {
    int i = blockIdx.x * blockDim.x + threadIdx.x;
    if (i < n) out[i] = f2bf_rne(in[i]);
}

// ---------------------------------------------------------------------------
// fp32 [R,C] -> bf16 transposed [C,R], 32x32 LDS tile (coalesced both sides).
// R, C multiples of 32. block (32,8), grid (C/32, R/32).
// ---------------------------------------------------------------------------
__global__ void Caspar_cvt_bf16_T(const float* __restrict__ in,
                                  unsigned short* __restrict__ out,
                                  int R, int C) {
    __shared__ unsigned short tile[32][33];
    const int c0 = blockIdx.x * 32;
    const int r0 = blockIdx.y * 32;
    const int tx = threadIdx.x;
#pragma unroll
    for (int i = threadIdx.y; i < 32; i += 8)
        tile[i][tx] = f2bf_rne(in[(size_t)(r0 + i) * C + (c0 + tx)]);
    __syncthreads();
#pragma unroll
    for (int i = threadIdx.y; i < 32; i += 8)
        out[(size_t)(c0 + i) * R + (r0 + tx)] = tile[tx][i];
}

__global__ void Caspar_zero_f32(float* __restrict__ p, int n) {
    int i = blockIdx.x * blockDim.x + threadIdx.x;
    if (i < n) p[i] = 0.0f;
}

// ---------------------------------------------------------------------------
// bf16 WMMA GEMM: C[M,N] = act(A[M,K] @ B[K,N] + bias)
// A row-major bf16 [M,K]; Bt = B TRANSPOSED, row-major bf16 [N,K]; C fp32.
// One wave computes a 16x64 strip (one A fragment reused across 4 N tiles;
// all fragment loads are contiguous -> global_load_b128).
// K mult of 32; M mult of 16; N mult of 64.
// ---------------------------------------------------------------------------
__global__ void Caspar_gemm_bf16(const unsigned short* __restrict__ A,
                                 const unsigned short* __restrict__ Bt,
                                 const float* __restrict__ bias,
                                 float* __restrict__ C,
                                 int M, int N, int K, int relu) {
    const int lane  = threadIdx.x;      // 0..31 (wave32)
    const int m     = lane & 15;
    const int half  = lane >> 4;        // 0 or 1
    const int tileN = blockIdx.x * 64;
    const int tileM = blockIdx.y * 16;

    const int aRow = tileM + m;
    const int kA0  = half * 16;         // A: K offset within 32-chunk
    const int kB0  = half * 16;         // B: K offset within 32-chunk

    v8f acc[4] = {};
    for (int k0 = 0; k0 < K; k0 += 32) {
        v16bf a;
        const unsigned short* Ap = A + (size_t)aRow * K + k0 + kA0;
#pragma unroll
        for (int j = 0; j < 8; ++j) {
            a[j]     = us2bf(Ap[j]);        // K = k0 + kA0 + j
            a[8 + j] = us2bf(Ap[8 + j]);    // K = k0 + kA0 + 8 + j
        }
#pragma unroll
        for (int nt = 0; nt < 4; ++nt) {
            const unsigned short* Bp =
                Bt + (size_t)(tileN + nt * 16 + m) * K + k0 + kB0;
            v16bf b;
#pragma unroll
            for (int j = 0; j < 16; ++j)
                b[j] = us2bf(Bp[j]);        // contiguous: 2x b128
            acc[nt] = __builtin_amdgcn_wmma_f32_16x16x32_bf16(
                          false, a, false, b, (short)0, acc[nt], false, false);
        }
    }

#pragma unroll
    for (int nt = 0; nt < 4; ++nt) {
#pragma unroll
        for (int r = 0; r < 8; ++r) {
            int row = tileM + half * 8 + r;
            int col = tileN + nt * 16 + m;
            float v = acc[nt][r];
            if (bias) v += bias[col];
            if (relu) v = fmaxf(v, 0.0f);
            C[(size_t)row * N + col] = v;
        }
    }
}

// ---------------------------------------------------------------------------
// LSTM recurrence, phase A: i-gate + candidate + new cell state.
//   pre_i = XI[x_t] + h@h2i + c@c2i   (XI already has bias bi folded in)
//   i = sigmoid(pre_i); w = tanh(XC[x_t] + h@h2c); c_new = i*w + (1-i)*c
// grid = (HD/64, 2 dirs), block = (64 outputs, 4 k-chunks)
// ---------------------------------------------------------------------------
__global__ void Caspar_lstm_phaseA(
    const float* __restrict__ XI_lr, const float* __restrict__ XC_lr,
    const float* __restrict__ XI_rl, const float* __restrict__ XC_rl,
    const float* __restrict__ h2i_lr, const float* __restrict__ c2i_lr,
    const float* __restrict__ h2c_lr,
    const float* __restrict__ h2i_rl, const float* __restrict__ c2i_rl,
    const float* __restrict__ h2c_rl,
    const float* __restrict__ h_all_lr, const float* __restrict__ h_all_rl,
    float* __restrict__ c_all_lr, float* __restrict__ c_all_rl, int t) {
    __shared__ float sh_h[HD];
    __shared__ float sh_c[HD];
    __shared__ float red_i[4][64];
    __shared__ float red_c[4][64];

    const int dir = blockIdx.y;
    const float* XI    = dir ? XI_rl  : XI_lr;
    const float* XC    = dir ? XC_rl  : XC_lr;
    const float* h2i   = dir ? h2i_rl : h2i_lr;
    const float* c2i   = dir ? c2i_rl : c2i_lr;
    const float* h2c   = dir ? h2c_rl : h2c_lr;
    const float* h_all = dir ? h_all_rl : h_all_lr;
    float*       c_all = dir ? c_all_rl : c_all_lr;
    const int xrow = dir ? (SEQ - 1 - t) : t;

    const int tid = threadIdx.y * 64 + threadIdx.x;
    for (int i = tid; i < HD; i += 256) {
        sh_h[i] = h_all[(size_t)t * HD + i];
        sh_c[i] = c_all[(size_t)t * HD + i];
    }
    __syncthreads();

    const int j  = blockIdx.x * 64 + threadIdx.x;
    const int kc = threadIdx.y;
    const float* Wi = h2i + j;
    const float* Wp = c2i + j;
    const float* Wc = h2c + j;
    float acc_i = 0.0f, acc_c = 0.0f;
    const int kbeg = kc * (HD / 4), kend = kbeg + (HD / 4);
    for (int kk = kbeg; kk < kend; kk += 64) {
        __builtin_prefetch(Wi + (size_t)(kk + 64) * HD, 0, 0);
        __builtin_prefetch(Wp + (size_t)(kk + 64) * HD, 0, 0);
        __builtin_prefetch(Wc + (size_t)(kk + 64) * HD, 0, 0);
#pragma unroll 4
        for (int k = kk; k < kk + 64; ++k) {
            float hk = sh_h[k], ck = sh_c[k];
            acc_i = fmaf(hk, Wi[(size_t)k * HD], acc_i);
            acc_i = fmaf(ck, Wp[(size_t)k * HD], acc_i);
            acc_c = fmaf(hk, Wc[(size_t)k * HD], acc_c);
        }
    }
    red_i[kc][threadIdx.x] = acc_i;
    red_c[kc][threadIdx.x] = acc_c;
    __syncthreads();
    if (kc == 0) {
        float si = red_i[0][threadIdx.x] + red_i[1][threadIdx.x] +
                   red_i[2][threadIdx.x] + red_i[3][threadIdx.x];
        float sc = red_c[0][threadIdx.x] + red_c[1][threadIdx.x] +
                   red_c[2][threadIdx.x] + red_c[3][threadIdx.x];
        float pre_i = XI[(size_t)xrow * HD + j] + si;
        float it = 1.0f / (1.0f + __expf(-pre_i));
        float wt = tanhf(XC[(size_t)xrow * HD + j] + sc);
        float cn = it * wt + (1.0f - it) * sh_c[j];
        c_all[(size_t)(t + 1) * HD + j] = cn;
    }
}

// ---------------------------------------------------------------------------
// LSTM recurrence, phase B: o-gate (peepholes NEW cell state) + hidden.
//   o = sigmoid(XO[x_t] + c_new@c2o + h@h2o); h_new = o * tanh(c_new)
// ---------------------------------------------------------------------------
__global__ void Caspar_lstm_phaseB(
    const float* __restrict__ XO_lr, const float* __restrict__ XO_rl,
    const float* __restrict__ h2o_lr, const float* __restrict__ c2o_lr,
    const float* __restrict__ h2o_rl, const float* __restrict__ c2o_rl,
    float* __restrict__ h_all_lr, float* __restrict__ h_all_rl,
    const float* __restrict__ c_all_lr, const float* __restrict__ c_all_rl,
    int t) {
    __shared__ float sh_h[HD];
    __shared__ float sh_cn[HD];
    __shared__ float red_o[4][64];

    const int dir = blockIdx.y;
    const float* XO    = dir ? XO_rl  : XO_lr;
    const float* h2o   = dir ? h2o_rl : h2o_lr;
    const float* c2o   = dir ? c2o_rl : c2o_lr;
    float*       h_all = dir ? h_all_rl : h_all_lr;
    const float* c_all = dir ? c_all_rl : c_all_lr;
    const int xrow = dir ? (SEQ - 1 - t) : t;

    const int tid = threadIdx.y * 64 + threadIdx.x;
    for (int i = tid; i < HD; i += 256) {
        sh_h[i]  = h_all[(size_t)t * HD + i];
        sh_cn[i] = c_all[(size_t)(t + 1) * HD + i];
    }
    __syncthreads();

    const int j  = blockIdx.x * 64 + threadIdx.x;
    const int kc = threadIdx.y;
    const float* Wo = h2o + j;
    const float* Wp = c2o + j;
    float acc_o = 0.0f;
    const int kbeg = kc * (HD / 4), kend = kbeg + (HD / 4);
    for (int kk = kbeg; kk < kend; kk += 64) {
        __builtin_prefetch(Wo + (size_t)(kk + 64) * HD, 0, 0);
        __builtin_prefetch(Wp + (size_t)(kk + 64) * HD, 0, 0);
#pragma unroll 4
        for (int k = kk; k < kk + 64; ++k) {
            acc_o = fmaf(sh_h[k],  Wo[(size_t)k * HD], acc_o);
            acc_o = fmaf(sh_cn[k], Wp[(size_t)k * HD], acc_o);
        }
    }
    red_o[kc][threadIdx.x] = acc_o;
    __syncthreads();
    if (kc == 0) {
        float so = red_o[0][threadIdx.x] + red_o[1][threadIdx.x] +
                   red_o[2][threadIdx.x] + red_o[3][threadIdx.x];
        float pre_o = XO[(size_t)xrow * HD + j] + so;
        float ot = 1.0f / (1.0f + __expf(-pre_o));
        h_all[(size_t)(t + 1) * HD + j] = ot * tanhf(sh_cn[j]);
    }
}

// ---------------------------------------------------------------------------
// Build ff_in (bf16): [t, 0:H] = h_lr[t], [t, H:2H] = h_rl at position t
// (rl scan step s = SEQ-1-t produced h_rl_all[s+1] = h_rl_all[SEQ-t]).
// ---------------------------------------------------------------------------
__global__ void Caspar_build_ffin(const float* __restrict__ h_lr_all,
                                  const float* __restrict__ h_rl_all,
                                  unsigned short* __restrict__ ffin) {
    int idx = blockIdx.x * blockDim.x + threadIdx.x;
    if (idx >= SEQ * 2 * HD) return;
    int t = idx / (2 * HD);
    int j = idx % (2 * HD);
    float v = (j < HD) ? h_lr_all[(size_t)(t + 1) * HD + j]
                       : h_rl_all[(size_t)(SEQ - t) * HD + (j - HD)];
    ffin[idx] = f2bf_rne(v);
}

// ---------------------------------------------------------------------------
extern "C" void kernel_launch(void* const* d_in, const int* in_sizes, int n_in,
                              void* d_out, int out_size, void* d_ws, size_t ws_size,
                              hipStream_t stream) {
    (void)in_sizes; (void)n_in; (void)out_size; (void)ws_size;
    // input order: x, lr_{x2i,h2i,c2i,bi,x2o,h2o,c2o,bo,x2c,h2c,bc}, rl_{...},
    //              ff_w, ff_b, head_w, head_b
    const float* x      = (const float*)d_in[0];
    const float* p_lr[11]; const float* p_rl[11];
    for (int i = 0; i < 11; ++i) { p_lr[i] = (const float*)d_in[1 + i];
                                   p_rl[i] = (const float*)d_in[12 + i]; }
    const float* ff_w   = (const float*)d_in[23];
    const float* ff_b   = (const float*)d_in[24];
    const float* head_w = (const float*)d_in[25];
    const float* head_b = (const float*)d_in[26];
    // LSTM param index: 0 x2i, 1 h2i, 2 c2i, 3 bi, 4 x2o, 5 h2o, 6 c2o, 7 bo,
    //                   8 x2c, 9 h2c, 10 bc

    size_t off = 0;
    auto alloc = [&](size_t bytes) -> char* {
        char* p = (char*)d_ws + off;
        off += (bytes + 255) & ~(size_t)255;
        return p;
    };
    unsigned short* Xbf      = (unsigned short*)alloc((size_t)SEQ * DIN * 2);
    unsigned short* Wxi[2], *Wxo[2], *Wxc[2];     // transposed: [HD, DIN] bf16
    for (int d = 0; d < 2; ++d) {
        Wxi[d] = (unsigned short*)alloc((size_t)DIN * HD * 2);
        Wxo[d] = (unsigned short*)alloc((size_t)DIN * HD * 2);
        Wxc[d] = (unsigned short*)alloc((size_t)DIN * HD * 2);
    }
    float* XI[2]; float* XO[2]; float* XC[2];
    for (int d = 0; d < 2; ++d) {
        XI[d] = (float*)alloc((size_t)SEQ * HD * 4);
        XO[d] = (float*)alloc((size_t)SEQ * HD * 4);
        XC[d] = (float*)alloc((size_t)SEQ * HD * 4);
    }
    float* h_all[2]; float* c_all[2];
    for (int d = 0; d < 2; ++d) {
        h_all[d] = (float*)alloc((size_t)(SEQ + 1) * HD * 4);
        c_all[d] = (float*)alloc((size_t)(SEQ + 1) * HD * 4);
    }
    unsigned short* ffin_bf  = (unsigned short*)alloc((size_t)SEQ * 2 * HD * 2);
    unsigned short* ffw_bf   = (unsigned short*)alloc((size_t)2 * HD * FF * 2);  // [FF, 2HD]
    float*          hidden   = (float*)alloc((size_t)SEQ * FF * 4);
    unsigned short* hid_bf   = (unsigned short*)alloc((size_t)SEQ * FF * 2);
    unsigned short* headw_bf = (unsigned short*)alloc((size_t)FF * NA * 2);      // [NA, FF]

    auto cvt = [&](const float* src, unsigned short* dst, int n) {
        Caspar_cvt_bf16<<<(n + 255) / 256, 256, 0, stream>>>(src, dst, n);
    };
    auto cvtT = [&](const float* src, unsigned short* dst, int R, int C) {
        Caspar_cvt_bf16_T<<<dim3(C / 32, R / 32), dim3(32, 8), 0, stream>>>(
            src, dst, R, C);
    };

    // --- Phase 0a: precision conversion (+ transpose for all B operands) ---
    cvt(x, Xbf, SEQ * DIN);
    const float* plist[2][3] = {{p_lr[0], p_lr[4], p_lr[8]},
                                {p_rl[0], p_rl[4], p_rl[8]}};
    for (int d = 0; d < 2; ++d) {
        cvtT(plist[d][0], Wxi[d], DIN, HD);
        cvtT(plist[d][1], Wxo[d], DIN, HD);
        cvtT(plist[d][2], Wxc[d], DIN, HD);
    }
    cvtT(ff_w, ffw_bf, 2 * HD, FF);
    cvtT(head_w, headw_bf, FF, NA);

    // --- zero initial h/c rows ---
    for (int d = 0; d < 2; ++d) {
        Caspar_zero_f32<<<(HD + 255) / 256, 256, 0, stream>>>(h_all[d], HD);
        Caspar_zero_f32<<<(HD + 255) / 256, 256, 0, stream>>>(c_all[d], HD);
    }

    // --- Phase 0b: input projections via WMMA (bias folded in) ---
    const float* blist[2][3] = {{p_lr[3], p_lr[7], p_lr[10]},
                                {p_rl[3], p_rl[7], p_rl[10]}};
    for (int d = 0; d < 2; ++d) {
        Caspar_gemm_bf16<<<dim3(HD / 64, SEQ / 16), 32, 0, stream>>>(
            Xbf, Wxi[d], blist[d][0], XI[d], SEQ, HD, DIN, 0);
        Caspar_gemm_bf16<<<dim3(HD / 64, SEQ / 16), 32, 0, stream>>>(
            Xbf, Wxo[d], blist[d][1], XO[d], SEQ, HD, DIN, 0);
        Caspar_gemm_bf16<<<dim3(HD / 64, SEQ / 16), 32, 0, stream>>>(
            Xbf, Wxc[d], blist[d][2], XC[d], SEQ, HD, DIN, 0);
    }

    // --- Phase 1: sequential recurrence, both directions per launch ---
    dim3 sgrid(HD / 64, 2), sblk(64, 4);
    for (int t = 0; t < SEQ; ++t) {
        Caspar_lstm_phaseA<<<sgrid, sblk, 0, stream>>>(
            XI[0], XC[0], XI[1], XC[1],
            p_lr[1], p_lr[2], p_lr[9],   // lr h2i, c2i, h2c
            p_rl[1], p_rl[2], p_rl[9],   // rl h2i, c2i, h2c
            h_all[0], h_all[1], c_all[0], c_all[1], t);
        Caspar_lstm_phaseB<<<sgrid, sblk, 0, stream>>>(
            XO[0], XO[1],
            p_lr[5], p_lr[6],            // lr h2o, c2o
            p_rl[5], p_rl[6],            // rl h2o, c2o
            h_all[0], h_all[1], c_all[0], c_all[1], t);
    }

    // --- Phase 2: FF + head ---
    Caspar_build_ffin<<<(SEQ * 2 * HD + 255) / 256, 256, 0, stream>>>(
        h_all[0], h_all[1], ffin_bf);
    Caspar_gemm_bf16<<<dim3(FF / 64, SEQ / 16), 32, 0, stream>>>(
        ffin_bf, ffw_bf, ff_b, hidden, SEQ, FF, 2 * HD, 1);
    cvt(hidden, hid_bf, SEQ * FF);
    Caspar_gemm_bf16<<<dim3(NA / 64, SEQ / 16), 32, 0, stream>>>(
        hid_bf, headw_bf, head_b, (float*)d_out, SEQ, NA, FF, 0);
}
